// CrossAttention_54769422959297
// MI455X (gfx1250) — matrix operand order repaired and checked by reading, not compiled, not run on hardware
//
#include <hip/hip_runtime.h>
#include <hip/hip_bf16.h>

typedef _Float16 v2h  __attribute__((ext_vector_type(2)));
typedef _Float16 v4h  __attribute__((ext_vector_type(4)));
typedef _Float16 v8h  __attribute__((ext_vector_type(8)));
typedef _Float16 v16h __attribute__((ext_vector_type(16)));
typedef float    v8f  __attribute__((ext_vector_type(8)));
typedef unsigned int u32x4 __attribute__((ext_vector_type(4)));
typedef int          i32x8 __attribute__((ext_vector_type(8)));
typedef int          i32x4 __attribute__((ext_vector_type(4)));

constexpr int BATCH  = 4;
constexpr int SEQ_Q  = 1024;
constexpr int SEQ_KV = 1024;
constexpr int DIM    = 512;   // DQ == DKV == INNER
constexpr int HEADS  = 8;
constexpr int HDIM   = 64;

// ---------------------------------------------------------------------------
// WMMA helpers (CDNA5 v_wmma_f32_16x16x32_f16, wave32)
// ---------------------------------------------------------------------------
__device__ __forceinline__ v8f wmma_f16(v16h a, v16h b, v8f c) {
  return __builtin_amdgcn_wmma_f32_16x16x32_f16(false, a, false, b, (short)0, c,
                                                false, false);
}

// A-matrix fragment (16x32 f16). ISA layout:
//  lanes 0-15 : row = lane,    elems 0-7 -> K 0..7,   elems 8-15 -> K 16..23
//  lanes 16-31: row = lane-16, elems 0-7 -> K 8..15,  elems 8-15 -> K 24..31
__device__ __forceinline__ v16h load_a_frag(const _Float16* base, int row0,
                                            int k0, int stride) {
  const int lane = threadIdx.x & 31;
  const int r = lane & 15, hi = lane >> 4;
  const _Float16* p = base + (size_t)(row0 + r) * stride + k0 + hi * 8;
  v8h lo = *(const v8h*)p;
  v8h hh = *(const v8h*)(p + 16);
  return __builtin_shufflevector(lo, hh, 0, 1, 2, 3, 4, 5, 6, 7, 8, 9, 10, 11,
                                 12, 13, 14, 15);
}

// B-matrix fragment (32x16 f16), source stored N-major: Bt[n][k].
//  lane: n = lane&15, elems 0-7 -> K = (lane>>4)*8 + e, elems 8-15 -> K = 16 + (lane>>4)*8 + (e-8)
__device__ __forceinline__ v16h load_b_frag(const _Float16* base, int n0,
                                            int k0, int stride) {
  const int lane = threadIdx.x & 31;
  const _Float16* p =
      base + (size_t)(n0 + (lane & 15)) * stride + k0 + ((lane >> 4) * 8);
  v8h lo = *(const v8h*)p;
  v8h hh = *(const v8h*)(p + 16);
  return __builtin_shufflevector(lo, hh, 0, 1, 2, 3, 4, 5, 6, 7, 8, 9, 10, 11,
                                 12, 13, 14, 15);
}

// C/D layout: element j of v8f <-> row = (lane>>4)*8 + j, col = lane&15

// ---------------------------------------------------------------------------
// Kernel 0: weight transpose + f32 -> f16 convert.  Wt[n*K + k] = W[k*N + n]
// ---------------------------------------------------------------------------
__global__ __launch_bounds__(256) void transpose_f16_kernel(
    const float* __restrict__ W, _Float16* __restrict__ Wt, int K, int N) {
  int idx = blockIdx.x * 256 + threadIdx.x;
  if (idx < K * N) {
    int n = idx / K;
    int k = idx - n * K;
    Wt[idx] = (_Float16)W[(size_t)k * N + n];
  }
}

// ---------------------------------------------------------------------------
// Kernel 1: Y[M,N] (f16) = oscale * (A[M,K] f32 @ Wt[N,K] f16)
// Block: 256 threads = 8 waves, tile 32(M) x 256(N); wave = 16x64.
// ---------------------------------------------------------------------------
__global__ __launch_bounds__(256) void gemm_f16_kernel(
    const float* __restrict__ A, const _Float16* __restrict__ Wt,
    _Float16* __restrict__ Y, int M, int N, int K, float oscale) {
  __shared__ __align__(32) _Float16 Alds[32 * 32];
  const int tid = threadIdx.x;
  const int wave = tid >> 5, lane = tid & 31;
  const int m0 = blockIdx.x * 32;
  const int n0 = blockIdx.y * 256 + (wave >> 1) * 64;
  const int mrow = (wave & 1) * 16;
  const int arow = tid >> 3, akp = (tid & 7) * 4;
  v8f c0 = {}, c1 = {}, c2 = {}, c3 = {};

  for (int kc = 0; kc < K; kc += 32) {
    __syncthreads();
    float4 f = *(const float4*)(A + (size_t)(m0 + arow) * K + kc + akp);
    v4h h4 = {(_Float16)f.x, (_Float16)f.y, (_Float16)f.z, (_Float16)f.w};
    *(v4h*)&Alds[arow * 32 + akp] = h4;
    __syncthreads();

    v16h a = load_a_frag(Alds, mrow, 0, 32);
    c0 = wmma_f16(a, load_b_frag(Wt, n0 + 0,  kc, K), c0);
    c1 = wmma_f16(a, load_b_frag(Wt, n0 + 16, kc, K), c1);
    c2 = wmma_f16(a, load_b_frag(Wt, n0 + 32, kc, K), c2);
    c3 = wmma_f16(a, load_b_frag(Wt, n0 + 48, kc, K), c3);
  }

  const int hi = lane >> 4, col = lane & 15;
  for (int j = 0; j < 8; ++j) {
    size_t row = (size_t)(m0 + mrow + hi * 8 + j);
    Y[row * N + n0 + 0  + col] = (_Float16)(c0[j] * oscale);
    Y[row * N + n0 + 16 + col] = (_Float16)(c1[j] * oscale);
    Y[row * N + n0 + 32 + col] = (_Float16)(c2[j] * oscale);
    Y[row * N + n0 + 48 + col] = (_Float16)(c3[j] * oscale);
  }
}

// ---------------------------------------------------------------------------
// Kernel 2: KV projection. K-half -> Kout[4096][512] row-major f16.
// V-half -> Vt[b][h][d][j] f16 (head-transposed), written as v8h (elem j of
// the C-layout walks the kv-row index, which is Vt's contiguous dim).
// ---------------------------------------------------------------------------
__global__ __launch_bounds__(256) void gemm_kv_kernel(
    const float* __restrict__ A, const _Float16* __restrict__ Wt,
    _Float16* __restrict__ Kout, _Float16* __restrict__ Vt) {
  __shared__ __align__(32) _Float16 Alds[32 * 32];
  const int tid = threadIdx.x;
  const int wave = tid >> 5, lane = tid & 31;
  const int m0 = blockIdx.x * 32;
  const int n0 = blockIdx.y * 256 + (wave >> 1) * 64;
  const int mrow = (wave & 1) * 16;
  const int arow = tid >> 3, akp = (tid & 7) * 4;
  v8f c0 = {}, c1 = {}, c2 = {}, c3 = {};

  for (int kc = 0; kc < DIM; kc += 32) {
    __syncthreads();
    float4 f = *(const float4*)(A + (size_t)(m0 + arow) * DIM + kc + akp);
    v4h h4 = {(_Float16)f.x, (_Float16)f.y, (_Float16)f.z, (_Float16)f.w};
    *(v4h*)&Alds[arow * 32 + akp] = h4;
    __syncthreads();

    v16h a = load_a_frag(Alds, mrow, 0, 32);
    c0 = wmma_f16(a, load_b_frag(Wt, n0 + 0,  kc, DIM), c0);
    c1 = wmma_f16(a, load_b_frag(Wt, n0 + 16, kc, DIM), c1);
    c2 = wmma_f16(a, load_b_frag(Wt, n0 + 32, kc, DIM), c2);
    c3 = wmma_f16(a, load_b_frag(Wt, n0 + 48, kc, DIM), c3);
  }

  const int hi = lane >> 4, col = lane & 15;
  const int rowbase = m0 + mrow + hi * 8;  // multiple of 8, same batch for j=0..7
  if (n0 < DIM) {  // K half: row-major [4096][512]
    for (int j = 0; j < 8; ++j) {
      size_t row = (size_t)(rowbase + j);
      Kout[row * DIM + n0 + 0  + col] = (_Float16)c0[j];
      Kout[row * DIM + n0 + 16 + col] = (_Float16)c1[j];
      Kout[row * DIM + n0 + 32 + col] = (_Float16)c2[j];
      Kout[row * DIM + n0 + 48 + col] = (_Float16)c3[j];
    }
  } else {  // V half: Vt[((b*8+h)*64+d)][j], contiguous over j
    const int b = rowbase >> 10;
    const int jj = rowbase & 1023;
    v8h h0, h1, h2, h3;
    for (int j = 0; j < 8; ++j) {
      h0[j] = (_Float16)c0[j]; h1[j] = (_Float16)c1[j];
      h2[j] = (_Float16)c2[j]; h3[j] = (_Float16)c3[j];
    }
    int np0 = n0 - DIM + col;
    *(v8h*)(Vt + ((size_t)(b * HEADS + ((np0 + 0)  >> 6)) * HDIM + ((np0 + 0)  & 63)) * SEQ_KV + jj) = h0;
    *(v8h*)(Vt + ((size_t)(b * HEADS + ((np0 + 16) >> 6)) * HDIM + ((np0 + 16) & 63)) * SEQ_KV + jj) = h1;
    *(v8h*)(Vt + ((size_t)(b * HEADS + ((np0 + 32) >> 6)) * HDIM + ((np0 + 32) & 63)) * SEQ_KV + jj) = h2;
    *(v8h*)(Vt + ((size_t)(b * HEADS + ((np0 + 48) >> 6)) * HDIM + ((np0 + 48) & 63)) * SEQ_KV + jj) = h3;
  }
}

// ---------------------------------------------------------------------------
// Kernel 3: flash attention, no in-loop staging: B-fragments straight from
// L2-resident K [4096][512] and Vt [b,h,d][j]. grid=(SEQ_Q/128, B*H), 256 thr.
// ---------------------------------------------------------------------------
__global__ __launch_bounds__(256) void attn_kernel(
    const _Float16* __restrict__ Qh, const _Float16* __restrict__ Kf,
    const _Float16* __restrict__ Vt, _Float16* __restrict__ O) {
  __shared__ __align__(32) _Float16 Plds[8][16 * 32];

  const int tid = threadIdx.x, wave = tid >> 5, lane = tid & 31;
  const int bh = blockIdx.y, b = bh >> 3, h = bh & 7;
  const int q0 = blockIdx.x * 128 + wave * 16;
  const int hi = lane >> 4, col = lane & 15;

  const _Float16* qbase = Qh + (size_t)(b * SEQ_Q + q0) * DIM + h * HDIM;
  v16h qf0 = load_a_frag(qbase, 0, 0, DIM);   // d = 0..31
  v16h qf1 = load_a_frag(qbase, 0, 32, DIM);  // d = 32..63

  const _Float16* kb = Kf + (size_t)b * SEQ_KV * DIM + h * HDIM;  // stride DIM
  const _Float16* vb = Vt + (size_t)bh * HDIM * SEQ_KV;           // stride SEQ_KV

  float m[8], l[8];
  v8f o0 = {}, o1 = {}, o2 = {}, o3 = {};
  for (int j = 0; j < 8; ++j) { m[j] = -3.0e38f; l[j] = 0.0f; }

  for (int j0 = 0; j0 < SEQ_KV; j0 += 32) {
    if (j0 + 32 < SEQ_KV) {  // prefetch next chunk (global_prefetch_b8)
      __builtin_prefetch(kb + (size_t)(j0 + 32 + lane) * DIM, 0, 1);
      __builtin_prefetch(vb + (size_t)lane * SEQ_KV + j0 + 32, 0, 1);
      __builtin_prefetch(vb + (size_t)(lane + 32) * SEQ_KV + j0 + 32, 0, 1);
    }

    // S = Q K^T
    v8f s0 = {}, s1 = {};
    s0 = wmma_f16(qf0, load_b_frag(kb, j0 + 0,  0,  DIM), s0);
    s0 = wmma_f16(qf1, load_b_frag(kb, j0 + 0,  32, DIM), s0);
    s1 = wmma_f16(qf0, load_b_frag(kb, j0 + 16, 0,  DIM), s1);
    s1 = wmma_f16(qf1, load_b_frag(kb, j0 + 16, 32, DIM), s1);

    // online softmax (rows live in 16-lane halves of the C layout)
    float corr[8];
    for (int j = 0; j < 8; ++j) {
      float v = fmaxf(s0[j], s1[j]);
      for (int msk = 1; msk < 16; msk <<= 1)
        v = fmaxf(v, __shfl_xor(v, msk, 32));
      float mc = fmaxf(m[j], v);
      corr[j] = __expf(m[j] - mc);
      float p0 = __expf(s0[j] - mc);
      float p1 = __expf(s1[j] - mc);
      s0[j] = p0;
      s1[j] = p1;
      float rsum = p0 + p1;
      for (int msk = 1; msk < 16; msk <<= 1) rsum += __shfl_xor(rsum, msk, 32);
      l[j] = l[j] * corr[j] + rsum;
      m[j] = mc;
    }
    for (int j = 0; j < 8; ++j) {
      o0[j] *= corr[j]; o1[j] *= corr[j];
      o2[j] *= corr[j]; o3[j] *= corr[j];
    }

    // P: C-layout -> A-layout through per-wave LDS scratch
    _Float16* pw = &Plds[wave][0];
    for (int j = 0; j < 8; ++j) {
      int r = hi * 8 + j;
      pw[r * 32 + col]      = (_Float16)s0[j];
      pw[r * 32 + 16 + col] = (_Float16)s1[j];
    }
    asm volatile("s_wait_dscnt 0" ::: "memory");
    v16h pf = load_a_frag(pw, 0, 0, 32);

    // O += P V
    o0 = wmma_f16(pf, load_b_frag(vb, 0,  j0, SEQ_KV), o0);
    o1 = wmma_f16(pf, load_b_frag(vb, 16, j0, SEQ_KV), o1);
    o2 = wmma_f16(pf, load_b_frag(vb, 32, j0, SEQ_KV), o2);
    o3 = wmma_f16(pf, load_b_frag(vb, 48, j0, SEQ_KV), o3);
  }

  _Float16* obase = O + (size_t)(b * SEQ_Q + q0) * DIM + h * HDIM;
  for (int j = 0; j < 8; ++j) {
    float inv = 1.0f / l[j];
    size_t r = (size_t)(hi * 8 + j) * DIM;
    obase[r + 0  + col] = (_Float16)(o0[j] * inv);
    obase[r + 16 + col] = (_Float16)(o1[j] * inv);
    obase[r + 32 + col] = (_Float16)(o2[j] * inv);
    obase[r + 48 + col] = (_Float16)(o3[j] * inv);
  }
}

// ---------------------------------------------------------------------------
// Kernel 4: out = LayerNorm(2*(X @ Wfc + bfc)) * gamma + beta   (f32 output)
// A tiles (already f16) DMA'd into LDS by the Tensor Data Mover (wave 0
// issues tensor_load_to_lds, waits on TENSORcnt, block barriers).
// ---------------------------------------------------------------------------
__global__ __launch_bounds__(256) void fc_ln_kernel(
    const _Float16* __restrict__ X, const _Float16* __restrict__ Wt,
    const float* __restrict__ bias, const float* __restrict__ gamma,
    const float* __restrict__ beta, float* __restrict__ out) {
  __shared__ __align__(32) _Float16 Alds[16 * 32];
  __shared__ float red[2][16];

  const int tid = threadIdx.x, wave = tid >> 5, lane = tid & 31;
  const int m0 = blockIdx.x * 16;
  const int n0 = wave * 64;
  const int hi = lane >> 4, col = lane & 15;
  v8f c0 = {}, c1 = {}, c2 = {}, c3 = {};

  const unsigned ldsAddr = (unsigned)(uintptr_t)(void*)&Alds[0];

  for (int kc = 0; kc < DIM; kc += 32) {
    __syncthreads();
    if (wave == 0) {
      // Tensor DMA descriptor: 2D tile, 32 (k, contiguous) x 16 (rows), f16,
      // tensor row stride 512, dest = Alds (packed 16x32).
      unsigned long long ga =
          (unsigned long long)(uintptr_t)(X + (size_t)m0 * DIM + kc);
      u32x4 g0 = {1u,                       // count=1, user mode
                  ldsAddr,                  // lds_addr
                  (unsigned)ga,             // global_addr[31:0]
                  (unsigned)((ga >> 32) & 0x1FFFFFFu) | (2u << 30)};  // type=2
      i32x8 g1 = {(int)(1u << 16),    // data_size = 2 bytes
                  (int)(32u << 16),   // tensor_dim0 = 32
                  (int)(16u << 16),   // tensor_dim1 = 16
                  (int)(32u << 16),   // tile_dim0 = 32
                  16,                 // tile_dim1 = 16, tile_dim2 = 0
                  512,                // tensor_dim0_stride = 512
                  0, 0};
      i32x4 gz4 = {0, 0, 0, 0};
      i32x8 gz8 = {0, 0, 0, 0, 0, 0, 0, 0};
      __builtin_amdgcn_tensor_load_to_lds(g0, g1, gz4, gz4, gz8, 0);
      __builtin_amdgcn_s_wait_tensorcnt(0);
    }
    __syncthreads();

    v16h a = load_a_frag(Alds, 0, 0, 32);
    c0 = wmma_f16(a, load_b_frag(Wt, n0 + 0,  kc, DIM), c0);
    c1 = wmma_f16(a, load_b_frag(Wt, n0 + 16, kc, DIM), c1);
    c2 = wmma_f16(a, load_b_frag(Wt, n0 + 32, kc, DIM), c2);
    c3 = wmma_f16(a, load_b_frag(Wt, n0 + 48, kc, DIM), c3);
  }

  // bias + "out += out" doubling
  float b0 = bias[n0 + col], b1 = bias[n0 + 16 + col];
  float b2 = bias[n0 + 32 + col], b3 = bias[n0 + 48 + col];
  for (int j = 0; j < 8; ++j) {
    c0[j] = 2.0f * (c0[j] + b0);
    c1[j] = 2.0f * (c1[j] + b1);
    c2[j] = 2.0f * (c2[j] + b2);
    c3[j] = 2.0f * (c3[j] + b3);
  }

  // cross-wave LayerNorm reduction over 512 cols
  if (tid < 16) { red[0][tid] = 0.0f; red[1][tid] = 0.0f; }
  __syncthreads();
  for (int j = 0; j < 8; ++j) {
    float s  = c0[j] + c1[j] + c2[j] + c3[j];
    float s2 = c0[j] * c0[j] + c1[j] * c1[j] + c2[j] * c2[j] + c3[j] * c3[j];
    for (int msk = 1; msk < 16; msk <<= 1) {
      s  += __shfl_xor(s, msk, 32);
      s2 += __shfl_xor(s2, msk, 32);
    }
    if (col == 0) {
      atomicAdd(&red[0][hi * 8 + j], s);
      atomicAdd(&red[1][hi * 8 + j], s2);
    }
  }
  __syncthreads();

  float g0 = gamma[n0 + col], g1 = gamma[n0 + 16 + col];
  float g2 = gamma[n0 + 32 + col], g3 = gamma[n0 + 48 + col];
  float e0 = beta[n0 + col], e1 = beta[n0 + 16 + col];
  float e2 = beta[n0 + 32 + col], e3 = beta[n0 + 48 + col];
  for (int j = 0; j < 8; ++j) {
    int r = hi * 8 + j;
    float mu   = red[0][r] * (1.0f / DIM);
    float var  = red[1][r] * (1.0f / DIM) - mu * mu;
    float rstd = rsqrtf(var + 1e-6f);
    float* orow = out + (size_t)(m0 + r) * DIM;
    orow[n0 + 0  + col] = (c0[j] - mu) * rstd * g0 + e0;
    orow[n0 + 16 + col] = (c1[j] - mu) * rstd * g1 + e1;
    orow[n0 + 32 + col] = (c2[j] - mu) * rstd * g2 + e2;
    orow[n0 + 48 + col] = (c3[j] - mu) * rstd * g3 + e3;
  }
}

// ---------------------------------------------------------------------------
// Launch
// ---------------------------------------------------------------------------
extern "C" void kernel_launch(void* const* d_in, const int* in_sizes, int n_in,
                              void* d_out, int out_size, void* d_ws,
                              size_t ws_size, hipStream_t stream) {
  const float* q     = (const float*)d_in[0];  // (4,1024,512)
  const float* kv    = (const float*)d_in[1];  // (4,1024,512)
  const float* Wq    = (const float*)d_in[2];  // (512,512)
  const float* Wkv   = (const float*)d_in[3];  // (512,1024)
  const float* Wfc   = (const float*)d_in[4];  // (512,512)
  const float* bfc   = (const float*)d_in[5];  // (512,)
  const float* gamma = (const float*)d_in[6];  // (512,)
  const float* beta  = (const float*)d_in[7];  // (512,)
  float* out = (float*)d_out;                  // (4,1024,512) f32

  // workspace layout (f16): Wq_t | Wkv_t | Wfc_t | qh | K | Vt | attn (~18 MB)
  _Float16* ws    = (_Float16*)d_ws;
  _Float16* Wq_t  = ws;
  _Float16* Wkv_t = Wq_t + (size_t)512 * 512;
  _Float16* Wfc_t = Wkv_t + (size_t)512 * 1024;
  _Float16* qh    = Wfc_t + (size_t)512 * 512;
  _Float16* Kbuf  = qh + (size_t)4096 * 512;
  _Float16* Vt    = Kbuf + (size_t)4096 * 512;
  _Float16* attn  = Vt + (size_t)4096 * 512;

  transpose_f16_kernel<<<(512 * 512) / 256, 256, 0, stream>>>(Wq, Wq_t, 512, 512);
  transpose_f16_kernel<<<(512 * 1024) / 256, 256, 0, stream>>>(Wkv, Wkv_t, 512, 1024);
  transpose_f16_kernel<<<(512 * 512) / 256, 256, 0, stream>>>(Wfc, Wfc_t, 512, 512);

  // qh = (q @ Wq) * DH^-0.5  (softmax scale folded into Q)
  gemm_f16_kernel<<<dim3(4096 / 32, 512 / 256), 256, 0, stream>>>(
      q, Wq_t, qh, 4096, 512, 512, 0.125f);
  // K row-major + V head-transposed, straight out of the projection
  gemm_kv_kernel<<<dim3(4096 / 32, 1024 / 256), 256, 0, stream>>>(kv, Wkv_t,
                                                                  Kbuf, Vt);

  attn_kernel<<<dim3(SEQ_Q / 128, BATCH * HEADS), 256, 0, stream>>>(qh, Kbuf,
                                                                    Vt, attn);

  fc_ln_kernel<<<(BATCH * SEQ_Q) / 16, 256, 0, stream>>>(attn, Wfc_t, bfc,
                                                         gamma, beta, out);
}
